// MultiHeadedAttention_3298534883607
// MI455X (gfx1250) — matrix operand order repaired and checked by reading, not compiled, not run on hardware
//
#include <hip/hip_runtime.h>
#include <hip/hip_bf16.h>

// Problem constants (B=4, G=6 -> BG=24)
#define NH      8          // heads
#define DK      64         // head dim
#define DM      512        // d_model
#define SEQ     1024       // sequence length
#define BG      24         // B*GROUPS
#define NHEADS  (BG*NH)    // 192
#define MROWS   (BG*SEQ)   // 24576

typedef __attribute__((ext_vector_type(16))) __bf16 v16bf;
typedef __attribute__((ext_vector_type(8)))  float  v8f;

union BF16FragU { v16bf v; unsigned short u[16]; unsigned int d[8]; };

// One-time stores: full round-to-nearest-even.
__device__ __forceinline__ unsigned short f32_to_bf16(float f) {
  union { float f; unsigned int u; } x; x.f = f;
  unsigned int r = x.u + 0x7FFFu + ((x.u >> 16) & 1u);
  return (unsigned short)(r >> 16);
}
__device__ __forceinline__ float bf16_to_f32(unsigned short h) {
  union { unsigned int u; float f; } x; x.u = ((unsigned int)h) << 16;
  return x.f;
}
// Hot-loop fragment conversion: pack two f32 -> bf16x2 with ONE v_perm_b32
// (truncation; error <= 2^-8 relative, irrelevant under softmax temp 1200).
__device__ __forceinline__ unsigned int bf16pack2(float lo, float hi) {
  return __builtin_amdgcn_perm(__float_as_uint(hi), __float_as_uint(lo),
                               0x07060302u);
}

// ---------------------------------------------------------------------------
// Kernel 0: zero the per-head column-sum accumulator (192*1024 floats)
// ---------------------------------------------------------------------------
__global__ void zero_kernel(float* p, int n) {
  int i = blockIdx.x * blockDim.x + threadIdx.x;
  if (i < n) p[i] = 0.0f;
}

// ---------------------------------------------------------------------------
// Kernel 1: projections  Y = X @ W^T + b  (M=24576, N=512, K=512), bf16 WMMA
// blockIdx.z selects {query,key,value}. Output stored bf16 in [head][s][d].
// Block = 256 threads = 8 waves; wave computes a 16(M) x 64(N) tile.
// ---------------------------------------------------------------------------
__global__ __launch_bounds__(256) void proj_kernel(
    const float* __restrict__ q_in, const float* __restrict__ k_in,
    const float* __restrict__ v_in,
    const float* __restrict__ Wq, const float* __restrict__ Wk,
    const float* __restrict__ Wv,
    const float* __restrict__ bq, const float* __restrict__ bk,
    const float* __restrict__ bv,
    unsigned short* __restrict__ qws, unsigned short* __restrict__ kws,
    unsigned short* __restrict__ vws)
{
  const int mat = blockIdx.z;
  const float* X    = (mat == 0) ? q_in : (mat == 1) ? k_in : v_in;
  const float* W    = (mat == 0) ? Wq   : (mat == 1) ? Wk   : Wv;
  const float* bias = (mat == 0) ? bq   : (mat == 1) ? bk   : bv;
  unsigned short* Y = (mat == 0) ? qws  : (mat == 1) ? kws  : vws;

  const int lane = threadIdx.x & 31;
  const int wave = threadIdx.x >> 5;
  const bool hi  = (lane >= 16);
  const int m0 = blockIdx.x * 64 + (wave >> 1) * 16;     // M tile base
  const int n0 = blockIdx.y * 128 + (wave & 1) * 64;     // N tile base
  const int row = m0 + (lane & 15);

  v8f acc[4] = { v8f{}, v8f{}, v8f{}, v8f{} };

  for (int k0 = 0; k0 < DM; k0 += 32) {
    // A fragment: 16x32 bf16. Lane l (<16): row M=l, K = k0+{0..7,16..23};
    // lane l+16: same row, K = k0+{8..15,24..31}.
    BF16FragU a;
    {
      const float* ap = X + (size_t)row * DM + k0 + (hi ? 8 : 0);
      #pragma unroll
      for (int j = 0; j < 4; ++j)
        a.d[j] = bf16pack2(ap[2 * j], ap[2 * j + 1]);
      #pragma unroll
      for (int j = 0; j < 4; ++j)
        a.d[4 + j] = bf16pack2(ap[16 + 2 * j], ap[16 + 2 * j + 1]);
    }
    // B fragments: B[k,n] = W[n,k] -> per lane: column n = lane&15,
    // K = k0+{0..7,16..23} (lanes>=16: k0+{8..15,24..31}), same as A.
    #pragma unroll
    for (int nn = 0; nn < 4; ++nn) {
      BF16FragU b;
      const int col = n0 + nn * 16 + (lane & 15);
      const float* bp = W + (size_t)col * DM + k0 + (hi ? 8 : 0);
      #pragma unroll
      for (int j = 0; j < 4; ++j)
        b.d[j] = bf16pack2(bp[2 * j], bp[2 * j + 1]);
      #pragma unroll
      for (int j = 0; j < 4; ++j)
        b.d[4 + j] = bf16pack2(bp[16 + 2 * j], bp[16 + 2 * j + 1]);
      acc[nn] = __builtin_amdgcn_wmma_f32_16x16x32_bf16(
          false, a.v, false, b.v, (short)0, acc[nn], false, false);
    }
  }

  // Store: C/D layout: VGPR r, lane -> M = m0 + r + (hi?8:0), N = n0+nn*16+(lane&15)
  #pragma unroll
  for (int nn = 0; nn < 4; ++nn) {
    const int n = n0 + nn * 16 + (lane & 15);
    const float bval = bias[n];
    const int h = n >> 6, d = n & 63;
    #pragma unroll
    for (int r = 0; r < 8; ++r) {
      const int m = m0 + r + (hi ? 8 : 0);
      const int bg = m >> 10, s = m & 1023;
      const size_t idx = ((size_t)((bg * NH + h) * SEQ + s)) * DK + d;
      Y[idx] = f32_to_bf16(acc[nn][r] + bval);
    }
  }
}

// ---------------------------------------------------------------------------
// Kernel 2: attention. Block = (head, 16-row q-tile). 8 waves split the 1024
// k-columns (128 each). All 8 score accumulators kept in registers; softmax
// (temperature 1/1200 -> |scores|<<1, max-subtraction safely skipped),
// p_attn written to d_out, column sums accumulated for the fused p@V mean.
// ---------------------------------------------------------------------------
__global__ __launch_bounds__(256) void attn_kernel(
    const unsigned short* __restrict__ qws,
    const unsigned short* __restrict__ kws,
    float* __restrict__ p_out, float* __restrict__ colsum)
{
  __shared__ float red[8 * 16];
  __shared__ float s_inv[16];

  const int head = blockIdx.y;             // 0..191
  const int q0   = blockIdx.x * 16;        // 0..1008
  const int lane = threadIdx.x & 31;
  const int wave = threadIdx.x >> 5;
  const bool hi  = (lane >= 16);

  const unsigned short* qh = qws + (size_t)head * SEQ * DK;
  const unsigned short* kh = kws + (size_t)head * SEQ * DK;

  // q fragments for d0 = 0, 32 (reused across all 8 column tiles)
  BF16FragU aq[2];
  {
    const int srow = q0 + (lane & 15);
    #pragma unroll
    for (int t = 0; t < 2; ++t) {
      const unsigned short* ap = qh + (size_t)srow * DK + t * 32 + (hi ? 8 : 0);
      #pragma unroll
      for (int i = 0; i < 8; ++i) aq[t].u[i]     = ap[i];
      #pragma unroll
      for (int i = 0; i < 8; ++i) aq[t].u[8 + i] = ap[16 + i];
    }
  }

  const float inv_temp = 1.0f / (8.0f * 150.0f);   // 1/sqrt(dk)/150

  v8f acc[8];
  #pragma unroll
  for (int ct = 0; ct < 8; ++ct) {
    acc[ct] = v8f{};
    const int krow = wave * 128 + ct * 16 + (lane & 15);
    #pragma unroll
    for (int t = 0; t < 2; ++t) {
      // B[d, j]: column j = lane&15 -> k-row krow, d = {0..7,16..23} (+8 for hi)
      BF16FragU b;
      const unsigned short* bp = kh + (size_t)krow * DK + t * 32 + (hi ? 8 : 0);
      #pragma unroll
      for (int i = 0; i < 8; ++i) b.u[i] = bp[i];
      #pragma unroll
      for (int i = 0; i < 8; ++i) b.u[8 + i] = bp[16 + i];
      acc[ct] = __builtin_amdgcn_wmma_f32_16x16x32_bf16(
          false, aq[t].v, false, b.v, (short)0, acc[ct], false, false);
    }
    // exp in place
    #pragma unroll
    for (int r = 0; r < 8; ++r) acc[ct][r] = __expf(acc[ct][r] * inv_temp);
  }

  // Row sums: per-lane partial over its 8 columns, reduce across 16 lanes.
  #pragma unroll
  for (int r = 0; r < 8; ++r) {
    float v = 0.0f;
    #pragma unroll
    for (int ct = 0; ct < 8; ++ct) v += acc[ct][r];
    v += __shfl_xor(v, 1);
    v += __shfl_xor(v, 2);
    v += __shfl_xor(v, 4);
    v += __shfl_xor(v, 8);
    if ((lane & 15) == 0) red[wave * 16 + r + (hi ? 8 : 0)] = v;
  }
  __syncthreads();
  if (threadIdx.x < 16) {
    float s = 0.0f;
    #pragma unroll
    for (int w = 0; w < 8; ++w) s += red[w * 16 + threadIdx.x];
    s_inv[threadIdx.x] = 1.0f / s;
  }
  __syncthreads();

  float inv_m[8];
  #pragma unroll
  for (int r = 0; r < 8; ++r) inv_m[r] = s_inv[r + (hi ? 8 : 0)];

  // Normalize, write p_attn, and fold column sums (for mean-over-q @ V).
  float* outp = p_out + ((size_t)head * SEQ + q0) * SEQ;
  #pragma unroll
  for (int ct = 0; ct < 8; ++ct) {
    const int col = wave * 128 + ct * 16 + (lane & 15);
    float cs = 0.0f;
    #pragma unroll
    for (int r = 0; r < 8; ++r) {
      const int m = r + (hi ? 8 : 0);
      const float p = acc[ct][r] * inv_m[r];
      outp[(size_t)m * SEQ + col] = p;
      cs += p;
    }
    cs += __shfl_xor(cs, 16);          // combine the two row-halves
    if (!hi) atomicAdd(&colsum[(size_t)head * SEQ + col], cs);
  }
}

// ---------------------------------------------------------------------------
// Kernel 3: out[h][d] = (1/S) * sum_k colsum[h][k]*v[h][k][d];
// x[c=d*8+h] = out[h][d] (torch transpose(0,1,3,2).reshape quirk);
// y[e] = sum_c x[c]*Wo[e,c] + bo[e].  One block per (b,g).
// ---------------------------------------------------------------------------
__global__ __launch_bounds__(256) void finish_kernel(
    const float* __restrict__ colsum, const unsigned short* __restrict__ vws,
    const float* __restrict__ Wo, const float* __restrict__ bo,
    float* __restrict__ y_out)
{
  __shared__ float x_s[DM];
  const int bg = blockIdx.x;  // 0..23

  for (int i = threadIdx.x; i < DM; i += 256) {
    const int h = i >> 6, d = i & 63;
    const int head = bg * NH + h;
    const float* cs = colsum + (size_t)head * SEQ;
    const unsigned short* vh = vws + (size_t)head * SEQ * DK + d;
    float s = 0.0f;
    for (int k = 0; k < SEQ; ++k) s += cs[k] * bf16_to_f32(vh[(size_t)k * DK]);
    x_s[d * NH + h] = s * (1.0f / (float)SEQ);
  }
  __syncthreads();
  for (int e = threadIdx.x; e < DM; e += 256) {
    const float* w = Wo + (size_t)e * DM;
    float s = bo[e];
    for (int c = 0; c < DM; ++c) s += x_s[c] * w[c];
    y_out[(size_t)bg * DM + e] = s;
  }
}

// ---------------------------------------------------------------------------
extern "C" void kernel_launch(void* const* d_in, const int* in_sizes, int n_in,
                              void* d_out, int out_size, void* d_ws, size_t ws_size,
                              hipStream_t stream) {
  const float* q_in = (const float*)d_in[0];
  const float* k_in = (const float*)d_in[1];
  const float* v_in = (const float*)d_in[2];
  const float* Wq = (const float*)d_in[3];
  const float* bq = (const float*)d_in[4];
  const float* Wk = (const float*)d_in[5];
  const float* bk = (const float*)d_in[6];
  const float* Wv = (const float*)d_in[7];
  const float* bv = (const float*)d_in[8];
  const float* Wo = (const float*)d_in[9];
  const float* bo = (const float*)d_in[10];

  float* y_out = (float*)d_out;                         // (B,G,1,512) = 12288
  float* p_out = (float*)d_out + (size_t)BG * DM;       // (B,G,H,S,S)

  // Workspace layout: q/k/v projections in bf16 ([head][s][d]) + colsum f32.
  const size_t proj_elems = (size_t)MROWS * DM;         // 12,582,912 per matrix
  unsigned short* qws = (unsigned short*)d_ws;
  unsigned short* kws = qws + proj_elems;
  unsigned short* vws = kws + proj_elems;
  float* colsum = (float*)(vws + proj_elems);           // 192*1024 floats

  // 0) zero colsum accumulator
  {
    const int n = NHEADS * SEQ;
    zero_kernel<<<(n + 255) / 256, 256, 0, stream>>>(colsum, n);
  }
  // 1) projections: grid (M/64, N/128, 3 matrices)
  proj_kernel<<<dim3(MROWS / 64, DM / 128, 3), 256, 0, stream>>>(
      q_in, k_in, v_in, Wq, Wk, Wv, bq, bk, bv, qws, kws, vws);
  // 2) attention: grid (q-tiles, heads)
  attn_kernel<<<dim3(SEQ / 16, NHEADS), 256, 0, stream>>>(qws, kws, p_out, colsum);
  // 3) colmean @ V, interleave, @ Wo^T
  finish_kernel<<<dim3(BG), 256, 0, stream>>>(colsum, vws, Wo, bo, y_out);
}
